// StackCounterDFRC_NN_38336878084697
// MI455X (gfx1250) — matrix operand order repaired and checked by reading, not compiled, not run on hardware
//
#include <hip/hip_runtime.h>
#include <hip/hip_bf16.h>
#include <math.h>

typedef __attribute__((ext_vector_type(16))) _Float16 v16h;
typedef __attribute__((ext_vector_type(8)))  float    v8f;

#define TT 32           // timesteps per chunk == one WMMA t-tile
#define THRESH 0.5f

__device__ __forceinline__ float delta_of(float push, float pop) {
    float d = 0.0f;
    if (push >= THRESH && push >= pop)      d = 1.0f;
    else if (pop >= THRESH && pop > push)   d = -1.0f;
    return d;
}

// Native CDNA5 v_tanh_f32 when available; branch-free exp fallback otherwise.
__device__ __forceinline__ float fast_tanh(float x) {
#if defined(__has_builtin)
#if __has_builtin(__builtin_amdgcn_tanhf)
    return __builtin_amdgcn_tanhf(x);
#define FAST_TANH_DONE 1
#endif
#endif
#ifndef FAST_TANH_DONE
    // tanh(|x|) = (1 - e^{-2|x|}) / (1 + e^{-2|x|}); e in (0,1], no overflow.
    float ax = fabsf(x);
    float e  = __expf(-2.0f * ax);
    float t  = (1.0f - e) * __builtin_amdgcn_rcpf(1.0f + e);
    return copysignf(t, x);
#endif
}

// ---------------- Pass 1: per-chunk delta totals ----------------
// thread = (chunk c, batch b); coalesced 8B/lane streaming reads of x.
__global__ __launch_bounds__(256) void pass1_chunk_sums(
        const float* __restrict__ x, float* __restrict__ chunkSum,
        int T, int B) {
    const int NC  = T / TT;
    const int idx = blockIdx.x * blockDim.x + threadIdx.x;
    if (idx >= NC * B) return;
    const int c = idx / B;
    const int b = idx - c * B;
    const float2* xp = reinterpret_cast<const float2*>(x);
    size_t base = (size_t)c * TT * B + b;
    float s = 0.0f;
#pragma unroll 4
    for (int t = 0; t < TT; ++t) {
        float2 v = xp[base + (size_t)t * B];
        s += delta_of(v.x, v.y);
    }
    chunkSum[idx] = s;
}

// ---------------- Pass 2: exclusive scan of chunk totals (tiny) ----------------
__global__ __launch_bounds__(256) void pass2_scan_chunks(
        const float* __restrict__ chunkSum, const float* __restrict__ state0,
        float* __restrict__ carryIn, int B, int NC) {
    const int b = blockIdx.x * blockDim.x + threadIdx.x;
    if (b >= B) return;
    float c = state0[2 * b];        // initial count from state0[b][0]
    for (int k = 0; k < NC; ++k) {
        carryIn[(size_t)k * B + b] = c;
        c += chunkSum[(size_t)k * B + b];
    }
}

// ---------------- Pass 3: WMMA triangular-matrix prefix sum + tanh ----------------
// One wave handles one (chunk, 16-batch-column) tile: P = Ltri(32x32) x D(32x16)
// computed as two 16x16x32 f16 WMMAs (A_top: k<=m, A_bot: k<=m+16).
__global__ __launch_bounds__(256) void pass3_wmma_scan(
        const float* __restrict__ x, const float* __restrict__ carryIn,
        float* __restrict__ out, int T, int B) {
    const int lane   = threadIdx.x & 31;
    const int wid    = blockIdx.x * (blockDim.x >> 5) + (threadIdx.x >> 5);
    const int NG     = B >> 4;            // column groups of 16
    const int NC     = T / TT;
    if (wid >= NC * NG) return;           // uniform per-wave: EXEC stays all-ones
    const int c      = wid / NG;
    const int g      = wid - c * NG;
    const int bcol   = (g << 4) + (lane & 15);
    const bool hi    = lane >= 16;
    const int M      = lane & 15;         // A-matrix row held by this lane

    // Constant triangular A matrices, per the 16-bit A 16x32 VGPR layout:
    //   low lanes:  VGPR v<4 -> K=2v+h      ; v>=4 -> K=8+2v+h
    //   high lanes: VGPR v<4 -> K=8+2v+h    ; v>=4 -> K=16+2v+h
    v16h Atop, Abot;
#pragma unroll
    for (int j = 0; j < 16; ++j) {
        const int v = j >> 1, h = j & 1;
        const int K = hi ? ((v < 4) ? (8 + 2 * v + h) : (16 + 2 * v + h))
                         : ((v < 4) ? (2 * v + h)     : (8 + 2 * v + h));
        Atop[j] = (K <= M)      ? (_Float16)1.0f : (_Float16)0.0f;
        Abot[j] = (K <= M + 16) ? (_Float16)1.0f : (_Float16)0.0f;
    }

    // B matrix = delta tile (32 t-rows x 16 b-cols), f16.
    // Layout: lane holds col N=lane&15; rows K = (hi?8:0)+j+(j>=8?8:0) for element j.
    const int tbase = c * TT;
    const float2* xp = reinterpret_cast<const float2*>(x);
    v16h Bm;
#pragma unroll
    for (int j = 0; j < 16; ++j) {
        const int r = (hi ? 8 : 0) + j + ((j >= 8) ? 8 : 0);
        float2 s = xp[(size_t)(tbase + r) * B + bcol];
        Bm[j] = (_Float16)delta_of(s.x, s.y);
    }

    const v8f zero = {};
    // P_top[m] = sum_{k<=m} D[k],  P_bot[m] = sum_{k<=m+16} D[k]
    v8f Ptop = __builtin_amdgcn_wmma_f32_16x16x32_f16(
        false, Atop, false, Bm, (short)0, zero, false, false);
    v8f Pbot = __builtin_amdgcn_wmma_f32_16x16x32_f16(
        false, Abot, false, Bm, (short)0, zero, false, false);

    const float carry = carryIn[(size_t)c * B + bcol];

    // C/D layout: VGPR r holds rows M = r (lanes 0-15) / M = 8+r (lanes 16-31).
    float2* op = reinterpret_cast<float2*>(out);
#pragma unroll
    for (int r = 0; r < 8; ++r) {
        const int t0 = tbase + (hi ? 8 : 0) + r;        // rows 0..15 of tile
        const float c0 = carry + Ptop[r];
        op[(size_t)t0 * B + bcol] = make_float2(c0, fast_tanh(c0));
        const int t1 = t0 + 16;                          // rows 16..31 of tile
        const float c1 = carry + Pbot[r];
        op[(size_t)t1 * B + bcol] = make_float2(c1, fast_tanh(c1));
    }
}

extern "C" void kernel_launch(void* const* d_in, const int* in_sizes, int n_in,
                              void* d_out, int out_size, void* d_ws, size_t ws_size,
                              hipStream_t stream) {
    const float* x      = (const float*)d_in[0];   // [T, B, 2]
    const float* state0 = (const float*)d_in[1];   // [B, 2]
    float* out = (float*)d_out;                    // [T, B, 2]

    const int B  = in_sizes[1] / 2;                // 4096
    const int T  = in_sizes[0] / (2 * B);          // 2048
    const int NC = T / TT;                         // 64 chunks

    float* chunkSum = (float*)d_ws;                // [NC, B]
    float* carryIn  = chunkSum + (size_t)NC * B;   // [NC, B]  (2 MiB total)

    // Pass 1: per-chunk delta totals (streams x through L2; x fits in 192MB L2)
    {
        int n = NC * B;
        pass1_chunk_sums<<<(n + 255) / 256, 256, 0, stream>>>(x, chunkSum, T, B);
    }
    // Pass 2: serial scan over NC=64 chunk totals per batch lane (tiny)
    pass2_scan_chunks<<<(B + 255) / 256, 256, 0, stream>>>(chunkSum, state0, carryIn, B, NC);
    // Pass 3: WMMA prefix-sum per tile + tanh + store (x re-read hits L2)
    {
        int waves  = NC * (B >> 4);                // 16384 waves
        int blocks = (waves + 7) / 8;              // 8 waves / 256-thread block
        pass3_wmma_scan<<<blocks, 256, 0, stream>>>(x, carryIn, out, T, B);
    }
}
